// Attention_10840497455152
// MI455X (gfx1250) — matrix operand order repaired and checked by reading, not compiled, not run on hardware
//
#include <hip/hip_runtime.h>
#include <hip/hip_bf16.h>

typedef __attribute__((ext_vector_type(8)))  float  v8f;
typedef __attribute__((ext_vector_type(16))) __bf16 v16bf;
typedef __attribute__((ext_vector_type(8)))  __bf16 v8bf;

#define WMMA_BF16(a, b, c) \
  __builtin_amdgcn_wmma_f32_16x16x32_bf16(false, (a), false, (b), (short)0, (c), false, false)

// ---- gfx1250 async global->LDS path (guarded so compile can't break) -------
#if __has_builtin(__builtin_amdgcn_global_load_async_to_lds_b128)
#define USE_ASYNC_LDS 1
typedef __attribute__((__vector_size__(4 * sizeof(int)))) int i32x4;
typedef __attribute__((address_space(1))) i32x4 g_i32x4;
typedef __attribute__((address_space(3))) i32x4 l_i32x4;
static __device__ __forceinline__ void async_b128(const void* g, void* l) {
  __builtin_amdgcn_global_load_async_to_lds_b128(
      (g_i32x4*)(unsigned long long)(g),
      (l_i32x4*)(unsigned int)(unsigned long long)(l), 0, 0);
}
static __device__ __forceinline__ void async_wait0() {
#if __has_builtin(__builtin_amdgcn_s_wait_asynccnt)
  __builtin_amdgcn_s_wait_asynccnt(0);
#else
  asm volatile("s_wait_asynccnt 0" ::: "memory");
#endif
}
static __device__ __forceinline__ void async_wait3() {
#if __has_builtin(__builtin_amdgcn_s_wait_asynccnt)
  __builtin_amdgcn_s_wait_asynccnt(3);
#else
  asm volatile("s_wait_asynccnt 3" ::: "memory");
#endif
}
#else
#define USE_ASYNC_LDS 0
#endif

static __device__ __forceinline__ unsigned short f32_to_bf16(float f) {
  unsigned int u = __float_as_uint(f);
  u += 0x7FFFu + ((u >> 16) & 1u);   // round-to-nearest-even
  return (unsigned short)(u >> 16);
}
static __device__ __forceinline__ float bf16_to_f32(unsigned short h) {
  return __uint_as_float(((unsigned int)h) << 16);
}
static __device__ __forceinline__ v16bf cat8(v8bf lo, v8bf hi) {
  return __builtin_shufflevector(lo, hi, 0, 1, 2, 3, 4, 5, 6, 7, 8, 9, 10, 11, 12, 13, 14, 15);
}

// ---- 16-lane row reductions in pure VALU via DPP row_ror (no LDS traffic) --
#if __has_builtin(__builtin_amdgcn_update_dpp)
template <int CTRL>
static __device__ __forceinline__ float dpp_f32(float x) {
  return __int_as_float(__builtin_amdgcn_update_dpp(0, __float_as_int(x), CTRL, 0xf, 0xf, true));
}
static __device__ __forceinline__ float rowmax16(float x) {
  x = fmaxf(x, dpp_f32<0x128>(x));  // row_ror:8
  x = fmaxf(x, dpp_f32<0x124>(x));  // row_ror:4
  x = fmaxf(x, dpp_f32<0x122>(x));  // row_ror:2
  x = fmaxf(x, dpp_f32<0x121>(x));  // row_ror:1
  return x;
}
static __device__ __forceinline__ float rowsum16(float x) {
  x += dpp_f32<0x128>(x);
  x += dpp_f32<0x124>(x);
  x += dpp_f32<0x122>(x);
  x += dpp_f32<0x121>(x);
  return x;
}
#else
static __device__ __forceinline__ float rowmax16(float x) {
  for (int off = 1; off < 16; off <<= 1) x = fmaxf(x, __shfl_xor(x, off, 32));
  return x;
}
static __device__ __forceinline__ float rowsum16(float x) {
  for (int off = 1; off < 16; off <<= 1) x += __shfl_xor(x, off, 32);
  return x;
}
#endif

// ---------------------------------------------------------------------------
// f32 -> bf16 bulk convert (4 elems / thread)
// ---------------------------------------------------------------------------
__global__ __launch_bounds__(256) void cvt_f32_bf16(const float* __restrict__ src,
                                                    unsigned short* __restrict__ dst, int n4) {
  int i = blockIdx.x * 256 + threadIdx.x;
  if (i >= n4) return;
  const float4 f = ((const float4*)src)[i];
  ushort4 o;
  o.x = f32_to_bf16(f.x); o.y = f32_to_bf16(f.y);
  o.z = f32_to_bf16(f.z); o.w = f32_to_bf16(f.w);
  ((ushort4*)dst)[i] = o;
}

// ---------------------------------------------------------------------------
// WMMA GEMM:  C[M,N] = A[M,K] * Bw[N,K]^T + bias
// block tile 128(M) x 64(N), K-step 32, double-buffered LDS with async
// global->LDS copies pipelined one tile ahead (s_wait_asynccnt 3 keeps the
// newest 3 copies in flight while WMMA consumes the older tile).
// MODE 0: bf16 scatter epilogue into q/k/v [B,H,N,D] buffers (bias fused)
// MODE 1: f32 row-major output with bias (final projection)
// ---------------------------------------------------------------------------
template <int MODE>
__global__ __launch_bounds__(256) void gemm_bf16_wmma(
    const unsigned short* __restrict__ A,   // [M][K] bf16
    const unsigned short* __restrict__ Bw,  // [N][K] bf16
    const float* __restrict__ bias,         // [N]
    unsigned short* __restrict__ q_out, unsigned short* __restrict__ k_out,
    unsigned short* __restrict__ v_out, float* __restrict__ c_out,
    int M, int N, int K) {
  __shared__ __align__(32) unsigned short As[2][128 * 32];  // 16 KB
  __shared__ __align__(32) unsigned short Bs[2][64 * 32];   //  8 KB
  const int tid = threadIdx.x;
  const int wave = tid >> 5, lane = tid & 31;
  const int half = lane >> 4, l15 = lane & 15;
  const int m0 = blockIdx.x * 128;
  const int n0 = blockIdx.y * 64;

  v8f acc[4];
#pragma unroll
  for (int t = 0; t < 4; ++t) acc[t] = (v8f)0.0f;

  const int a_row = tid >> 1, a_c = (tid & 1) * 16;   // 128 rows x 2x16-elem chunks
  const int b_row = tid >> 2, b_c = (tid & 3) * 8;    // 64 rows x 4x8-elem chunks
  const unsigned short* a_src = A + (size_t)(m0 + a_row) * K + a_c;
  const unsigned short* b_src = Bw + (size_t)(n0 + b_row) * K + b_c;
  const int a_off = a_row * 32 + a_c;
  const int b_off = b_row * 32 + b_c;

  // per-thread tile copy (3 x B128)
  auto copy_tile = [&](int bufi, int kt) {
#if USE_ASYNC_LDS
    async_b128(a_src + kt, &As[bufi][a_off]);
    async_b128(a_src + kt + 8, &As[bufi][a_off + 8]);
    async_b128(b_src + kt, &Bs[bufi][b_off]);
#else
    *(uint4*)(&As[bufi][a_off])     = *(const uint4*)(a_src + kt);
    *(uint4*)(&As[bufi][a_off + 8]) = *(const uint4*)(a_src + kt + 8);
    *(uint4*)(&Bs[bufi][b_off])     = *(const uint4*)(b_src + kt);
#endif
  };

  copy_tile(0, 0);                 // prologue: tile 0 -> buffer 0
  int buf = 0;
  for (int kt = 0; kt < K; kt += 32) {
    const int nxt = buf ^ 1;
    const bool have_next = (kt + 32) < K;
    if (have_next) {
      copy_tile(nxt, kt + 32);     // overwrite buffer whose readers hit last barrier
      __builtin_prefetch(a_src + kt + 64, 0, 0);
    }
#if USE_ASYNC_LDS
    if (have_next) async_wait3();  // wait for current tile only; next stays in flight
    else           async_wait0();
#endif
    __syncthreads();               // all waves' current-tile copies visible

    // A fragment: lane half 0 -> K 0..15, half 1 -> K 16..31 (contiguous)
    const unsigned short* ap = &As[buf][(wave * 16 + l15) * 32 + half * 16];
    const v16bf afrag = *(const v16bf*)ap;
#pragma unroll
    for (int t = 0; t < 4; ++t) {
      // B fragment: col n = t*16 + l15, K = half*16 .. half*16+15 (contiguous)
      const v16bf bfrag = *(const v16bf*)(&Bs[buf][(t * 16 + l15) * 32 + half * 16]);
      acc[t] = WMMA_BF16(afrag, bfrag, acc[t]);
    }
    __syncthreads();               // readers done before this buffer is overwritten
    buf = nxt;
  }

#pragma unroll
  for (int t = 0; t < 4; ++t) {
#pragma unroll
    for (int r = 0; r < 8; ++r) {
      const int mg = m0 + wave * 16 + half * 8 + r;   // C layout: M = r (+8 upper half)
      const int ng = n0 + t * 16 + l15;               // N = lane&15
      const float val = acc[t][r] + bias[ng];
      if (MODE == 0) {
        const int which = ng >> 10;
        const int h = (ng >> 6) & 15;
        const int d = ng & 63;
        const int b = mg >> 10;
        const int seq = mg & 1023;
        unsigned short* dst = (which == 0) ? q_out : ((which == 1) ? k_out : v_out);
        dst[((size_t)((b << 4) + h) * 1024 + seq) * 64 + d] = f32_to_bf16(val);
      } else {
        c_out[(size_t)mg * N + ng] = val;
      }
    }
  }
}

// ---------------------------------------------------------------------------
// RMS-norm over contiguous 64-elem rows (bf16 in/out, f32 math).
// One row per 16-lane half-wave, 4 elems/lane, pure DPP reduction (no LDS).
// ---------------------------------------------------------------------------
__global__ __launch_bounds__(256) void rmsnorm_bf16(unsigned short* __restrict__ buf,
                                                    const float* __restrict__ w, int nrows) {
  const int gh = (blockIdx.x * 256 + threadIdx.x) >> 4;  // global 16-lane group = row
  const int l16 = threadIdx.x & 15;
  if (gh >= nrows) return;
  unsigned short* row = buf + (size_t)gh * 64;
  uint2 u = *(const uint2*)(row + l16 * 4);
  float a0 = bf16_to_f32((unsigned short)(u.x & 0xFFFFu));
  float a1 = bf16_to_f32((unsigned short)(u.x >> 16));
  float a2 = bf16_to_f32((unsigned short)(u.y & 0xFFFFu));
  float a3 = bf16_to_f32((unsigned short)(u.y >> 16));
  float ss = a0 * a0 + a1 * a1 + a2 * a2 + a3 * a3;
  ss = rowsum16(ss);
  const float inv = rsqrtf(ss * (1.0f / 64.0f) + 1e-6f);
  a0 *= inv * w[l16 * 4 + 0];
  a1 *= inv * w[l16 * 4 + 1];
  a2 *= inv * w[l16 * 4 + 2];
  a3 *= inv * w[l16 * 4 + 3];
  u.x = (unsigned int)f32_to_bf16(a0) | ((unsigned int)f32_to_bf16(a1) << 16);
  u.y = (unsigned int)f32_to_bf16(a2) | ((unsigned int)f32_to_bf16(a3) << 16);
  *(uint2*)(row + l16 * 4) = u;
}

// ---------------------------------------------------------------------------
// Flash attention. grid (qblock, b*16+h). 128 query rows / block, 8 waves,
// wave owns 16 rows. Key blocks of 64, online softmax, WMMA for QK^T and PV.
// ---------------------------------------------------------------------------
__global__ __launch_bounds__(256) void flash_attn_bf16(
    const unsigned short* __restrict__ Q, const unsigned short* __restrict__ Kb,
    const unsigned short* __restrict__ Vb, unsigned short* __restrict__ Out) {
  __shared__ __align__(32) unsigned short Qs[128 * 64];     // 16 KB  [row][d]
  __shared__ __align__(32) unsigned short Ks[64 * 64];      //  8 KB  [key][d]
  __shared__ __align__(32) unsigned short Vt[64 * 64];      //  8 KB  [d][key] (transposed)
  __shared__ __align__(32) unsigned short Ps[8 * 16 * 64];  // 16 KB  per-wave P tile
  const int tid = threadIdx.x;
  const int wave = tid >> 5, lane = tid & 31;
  const int half = lane >> 4, l15 = lane & 15;
  const int bh = blockIdx.y;          // b*16 + h
  const int q0 = blockIdx.x * 128;
  const unsigned short* Qg = Q + (size_t)bh * 1024 * 64;
  const unsigned short* Kg = Kb + (size_t)bh * 1024 * 64;
  const unsigned short* Vg = Vb + (size_t)bh * 1024 * 64;

  // cooperative Q tile load: 128x64 bf16
  for (int i = tid; i < 1024; i += 256) {
    const int row = i >> 3, c = (i & 7) * 8;
#if USE_ASYNC_LDS
    async_b128(Qg + (size_t)(q0 + row) * 64 + c, Qs + row * 64 + c);
#else
    *(uint4*)(Qs + row * 64 + c) = *(const uint4*)(Qg + (size_t)(q0 + row) * 64 + c);
#endif
  }

  float m_r[8], l_r[8];
  v8f o[4];
#pragma unroll
  for (int t = 0; t < 4; ++t) o[t] = (v8f)0.0f;
#pragma unroll
  for (int r = 0; r < 8; ++r) { m_r[r] = -3.0e38f; l_r[r] = 0.0f; }

  unsigned short* pw = Ps + wave * (16 * 64);

  for (int s0 = 0; s0 < 1024; s0 += 64) {
    __syncthreads();   // Q ready (iter 0) / previous K,V consumers done
    for (int i = tid; i < 512; i += 256) {
      const int row = i >> 3, c = (i & 7) * 8;
#if USE_ASYNC_LDS
      async_b128(Kg + (size_t)(s0 + row) * 64 + c, Ks + row * 64 + c);
#else
      *(uint4*)(Ks + row * 64 + c) = *(const uint4*)(Kg + (size_t)(s0 + row) * 64 + c);
#endif
    }
    // V loaded transposed so PV B-fragments are contiguous LDS reads
    for (int i = tid; i < 2048; i += 256) {
      const int key = i >> 5;
      const int d2 = (i & 31) * 2;
      const unsigned int u = *(const unsigned int*)(Vg + (size_t)(s0 + key) * 64 + d2);
      Vt[d2 * 64 + key] = (unsigned short)(u & 0xFFFFu);
      Vt[(d2 + 1) * 64 + key] = (unsigned short)(u >> 16);
    }
#if USE_ASYNC_LDS
    async_wait0();     // covers K tile (and Q on first iteration)
#endif
    __syncthreads();

    // S = Q K^T  (16 rows x 64 keys per wave), K-dim = D = 64 -> 2 wmma steps
    v8f s[4];
#pragma unroll
    for (int t = 0; t < 4; ++t) s[t] = (v8f)0.0f;
#pragma unroll
    for (int ks = 0; ks < 2; ++ks) {
      const unsigned short* ap = Qs + (wave * 16 + l15) * 64 + ks * 32 + half * 16;
      const v16bf afrag = *(const v16bf*)ap;
#pragma unroll
      for (int t = 0; t < 4; ++t) {
        const v16bf bfrag = *(const v16bf*)(Ks + (t * 16 + l15) * 64 + ks * 32 + half * 16);
        s[t] = WMMA_BF16(afrag, bfrag, s[t]);
      }
    }

    // online softmax; lane holds col key=t*16+l15, rows r(+8 upper half)
#pragma unroll
    for (int r = 0; r < 8; ++r) {
      float mx = -3.0e38f;
#pragma unroll
      for (int t = 0; t < 4; ++t) { s[t][r] *= 0.125f; mx = fmaxf(mx, s[t][r]); }
      mx = rowmax16(mx);
      const float mnew = fmaxf(m_r[r], mx);
      const float fo = __expf(m_r[r] - mnew);
      m_r[r] = mnew;
      float ps = 0.0f;
#pragma unroll
      for (int t = 0; t < 4; ++t) {
        const float p = __expf(s[t][r] - mnew);
        s[t][r] = p;
        ps += p;
      }
      ps = rowsum16(ps);
      l_r[r] = l_r[r] * fo + ps;
#pragma unroll
      for (int t = 0; t < 4; ++t) o[t][r] *= fo;
    }

    // C-layout P -> LDS -> A-fragment layout (bf16)
#pragma unroll
    for (int t = 0; t < 4; ++t)
#pragma unroll
      for (int r = 0; r < 8; ++r)
        pw[(half * 8 + r) * 64 + t * 16 + l15] = f32_to_bf16(s[t][r]);

    // O += P @ V   (keys are the K-dim, 2 wmma steps of 32 keys)
#pragma unroll
    for (int ks = 0; ks < 2; ++ks) {
      const unsigned short* ap = pw + l15 * 64 + ks * 32 + half * 16;
      const v16bf afrag = *(const v16bf*)ap;
#pragma unroll
      for (int t = 0; t < 4; ++t) {
        const v16bf bfrag = *(const v16bf*)(Vt + (t * 16 + l15) * 64 + ks * 32 + half * 16);
        o[t] = WMMA_BF16(afrag, bfrag, o[t]);
      }
    }
  }

  // epilogue: normalize, write bf16 attn output in [B,N,C] layout
  const int b = bh >> 4, h = bh & 15;
#pragma unroll
  for (int t = 0; t < 4; ++t) {
#pragma unroll
    for (int r = 0; r < 8; ++r) {
      const int seq = q0 + wave * 16 + half * 8 + r;
      const int d = t * 16 + l15;
      const float val = o[t][r] / l_r[r];
      Out[(size_t)(b * 1024 + seq) * 1024 + h * 64 + d] = f32_to_bf16(val);
    }
  }
}

// ---------------------------------------------------------------------------
extern "C" void kernel_launch(void* const* d_in, const int* in_sizes, int n_in,
                              void* d_out, int out_size, void* d_ws, size_t ws_size,
                              hipStream_t stream) {
  const float* x      = (const float*)d_in[0];
  const float* w_qkv  = (const float*)d_in[1];
  const float* b_qkv  = (const float*)d_in[2];
  const float* qn_w   = (const float*)d_in[3];
  const float* kn_w   = (const float*)d_in[4];
  const float* w_proj = (const float*)d_in[5];
  const float* b_proj = (const float*)d_in[6];
  float* out = (float*)d_out;

  const size_t M1 = 1024ull * 1024ull;
  unsigned short* ws     = (unsigned short*)d_ws;
  unsigned short* xb     = ws;               // 8M elems (dead after GEMM1)
  unsigned short* wqkvb  = ws + 8 * M1;      // 3M
  unsigned short* wprojb = ws + 11 * M1;     // 1M
  unsigned short* qb     = ws + 12 * M1;     // 8M  [B,H,N,D]
  unsigned short* kbuf   = ws + 20 * M1;     // 8M
  unsigned short* vbuf   = ws + 28 * M1;     // 8M
  unsigned short* attnb  = xb;               // reuse: 8M [B,N,C]

  cvt_f32_bf16<<<(int)(8 * M1 / 4 / 256), 256, 0, stream>>>(x, xb, (int)(8 * M1 / 4));
  cvt_f32_bf16<<<(int)(3 * M1 / 4 / 256), 256, 0, stream>>>(w_qkv, wqkvb, (int)(3 * M1 / 4));
  cvt_f32_bf16<<<(int)(M1 / 4 / 256), 256, 0, stream>>>(w_proj, wprojb, (int)(M1 / 4));

  // QKV projection: [8192,1024] x [3072,1024]^T -> bf16 q/k/v [B,H,N,D]
  gemm_bf16_wmma<0><<<dim3(8192 / 128, 3072 / 64), 256, 0, stream>>>(
      xb, wqkvb, b_qkv, qb, kbuf, vbuf, nullptr, 8192, 3072, 1024);

  // per-head RMS norm of q and k (131072 rows of 64 each; 16 threads/row)
  rmsnorm_bf16<<<131072 * 16 / 256, 256, 0, stream>>>(qb, qn_w, 131072);
  rmsnorm_bf16<<<131072 * 16 / 256, 256, 0, stream>>>(kbuf, kn_w, 131072);

  // flash attention over all (b,h) pairs
  flash_attn_bf16<<<dim3(8, 128), 256, 0, stream>>>(qb, kbuf, vbuf, attnb);

  // output projection: [8192,1024] x [1024,1024]^T -> f32 out
  gemm_bf16_wmma<1><<<dim3(8192 / 128, 1024 / 64), 256, 0, stream>>>(
      attnb, wprojb, b_proj, nullptr, nullptr, nullptr, out, 8192, 1024, 1024);
}
